// NAND_41085657154023
// MI455X (gfx1250) — matrix operand order repaired and checked by reading, not compiled, not run on hardware
//
#include <hip/hip_runtime.h>
#include <hip/hip_bf16.h>

// ---------------------------------------------------------------------------
// Problem constants (match reference)
// ---------------------------------------------------------------------------
#define VEC_SIZE   256
#define N_VECS     8
#define N_CHOICES  2048
#define BATCH      2048
#define ROW_ELEMS  (N_VECS * VEC_SIZE)   // 2048 halfs per row of qb/wb

// Workgroup tiling: 8 waves, each owning a 16(b)x64(c) strip -> WG = 128x64.
// All 8 waves share one 64-column B panel staged in LDS.
#define COLS_PER_WG 64
#define B_STRIDE    264                  // 256 K-halfs + 8 pad (bank spread)

typedef __bf16 bf16x8  __attribute__((ext_vector_type(8)));
typedef __bf16 bf16x16 __attribute__((ext_vector_type(16)));
typedef float  v8f     __attribute__((ext_vector_type(8)));
typedef float  f32x4   __attribute__((ext_vector_type(4)));

// ---------------------------------------------------------------------------
// Pass 1: L2-normalize every 256-float vector, emit bf16.
// One wave32 per vector: 8 floats/lane, butterfly reduction (wave32 shuffles).
// ---------------------------------------------------------------------------
__global__ __launch_bounds__(256) void nand_norm_bf16(
    const float* __restrict__ q, const float* __restrict__ w,
    __bf16* __restrict__ qb, __bf16* __restrict__ wb) {
  const int wave = (int)((blockIdx.x * blockDim.x + threadIdx.x) >> 5);
  const int lane = (int)(threadIdx.x & 31);
  const int nQVecs = BATCH * N_VECS;                 // 16384 query vectors
  const float* src;
  __bf16* dst;
  if (wave < nQVecs) {
    src = q + (size_t)wave * VEC_SIZE;
    dst = qb + (size_t)wave * VEC_SIZE;
  } else {
    const int v = wave - nQVecs;
    src = w + (size_t)v * VEC_SIZE;
    dst = wb + (size_t)v * VEC_SIZE;
  }
  const f32x4* s4 = (const f32x4*)(src + lane * 8);
  f32x4 a = s4[0], b = s4[1];
  float ss = a[0]*a[0] + a[1]*a[1] + a[2]*a[2] + a[3]*a[3]
           + b[0]*b[0] + b[1]*b[1] + b[2]*b[2] + b[3]*b[3];
#pragma unroll
  for (int m = 16; m >= 1; m >>= 1) ss += __shfl_xor(ss, m, 32);
  const float inv = 1.0f / fmaxf(sqrtf(ss), 1e-6f);
  bf16x8 o;
#pragma unroll
  for (int i = 0; i < 4; ++i) o[i]     = (__bf16)(a[i] * inv);
#pragma unroll
  for (int i = 0; i < 4; ++i) o[4 + i] = (__bf16)(b[i] * inv);
  *(bf16x8*)(dst + lane * 8) = o;
}

// ---------------------------------------------------------------------------
// Pass 2: WMMA contraction + fused NAND epilogue, B panel staged in LDS.
//
// Grid: 512 WGs of 256 threads (8 waves). WG (bGroup, cGrp) covers rows
// [bGroup*128, +128) x cols [cGrp*64, +64). Wave w owns b-tile bGroup*8+w.
// Per n (K-chunk of 256): B panel (64 cols x 256 halfs = 32 KB) is
// prefetched global->VGPR during the previous chunk's compute, then
// barrier + ds_store + barrier. A rows are per-wave-unique -> direct global.
// ---------------------------------------------------------------------------
__global__ __launch_bounds__(256) void nand_wmma_kernel(
    const __bf16* __restrict__ qb, const __bf16* __restrict__ wb,
    const float* __restrict__ nandw, float* __restrict__ out) {
  __shared__ __align__(16) __bf16 shB[COLS_PER_WG * B_STRIDE];

  const int tid   = (int)threadIdx.x;
  const int lane  = tid & 31;
  const int wv    = tid >> 5;           // wave-in-WG: 0..7
  const int wg    = (int)blockIdx.x;
  const int bGrp  = wg >> 5;            // 0..15  (BATCH/128)
  const int cGrp  = wg & 31;            // 0..31  (N_CHOICES/64)
  const int bTile = bGrp * 8 + wv;      // 0..127
  const int cBase = cGrp << 6;

  const int mLane = lane & 15;          // row (A) / col (B,C,D) within tile
  const int hi    = lane >> 4;          // K half-group held by this lane

  // ---- B staging assignment: 4 threads per column, 8 x 16B each ----
  const int sCol = tid >> 2;            // 0..63
  const int sSeg = tid & 3;             // 0..3
  const __bf16* wSrc = wb + (size_t)(cBase + sCol) * ROW_ELEMS + sSeg * 8;
  __bf16* wDst = &shB[sCol * B_STRIDE + sSeg * 8];

  // ---- compute-side pointers ----
  const __bf16* aRow = qb + (size_t)((bTile << 4) + mLane) * ROW_ELEMS;
  int cIdx[4];
  const __bf16* shCol[4];
#pragma unroll
  for (int t = 0; t < 4; ++t) {
    cIdx[t]  = cBase + t * 16 + mLane;
    shCol[t] = &shB[(t * 16 + mLane) * B_STRIDE + hi * 16];
  }

  const v8f vzero = {0.f, 0.f, 0.f, 0.f, 0.f, 0.f, 0.f, 0.f};
  float prod[4][8];
#pragma unroll
  for (int t = 0; t < 4; ++t)
#pragma unroll
    for (int r = 0; r < 8; ++r) prod[t][r] = 1.0f;

  // Prefetch stage 0 into VGPRs, store to LDS, sync.
  bf16x8 pre[8];
#pragma unroll
  for (int i = 0; i < 8; ++i) pre[i] = *(const bf16x8*)(wSrc + i * 32);
#pragma unroll
  for (int i = 0; i < 8; ++i) *(bf16x8*)(wDst + i * 32) = pre[i];
  __syncthreads();

#pragma unroll 1
  for (int n = 0; n < N_VECS; ++n) {
    // Prefetch next stage global->VGPR while computing from LDS.
    if (n < N_VECS - 1) {
      const __bf16* src = wSrc + (n + 1) * VEC_SIZE;
#pragma unroll
      for (int i = 0; i < 8; ++i) pre[i] = *(const bf16x8*)(src + i * 32);
    }

    v8f acc[4] = {vzero, vzero, vzero, vzero};
    const int nOff = n << 8;            // n * VEC_SIZE

#pragma unroll
    for (int k = 0; k < 8; ++k) {       // K = 256 in steps of 32
      const int kOff = nOff + (k << 5);
      // A k order used here (contiguous, same as B): hi=0 holds K {0..15},
      // hi=1 holds K {16..31} of row mLane.
      bf16x8 a0 = *(const bf16x8*)(aRow + kOff + hi * 16);
      bf16x8 a1 = *(const bf16x8*)(aRow + kOff + hi * 16 + 8);
      bf16x16 A = __builtin_shufflevector(a0, a1,
          0, 1, 2, 3, 4, 5, 6, 7, 8, 9, 10, 11, 12, 13, 14, 15);
#pragma unroll
      for (int t = 0; t < 4; ++t) {
        // B k order used here (contiguous, same as A): lane hi-group holds K 16*hi..16*hi+15
        // of column mLane; two 16B LDS reads.
        bf16x8 b0 = *(const bf16x8*)(shCol[t] + (k << 5));
        bf16x8 b1 = *(const bf16x8*)(shCol[t] + (k << 5) + 8);
        bf16x16 Bm = __builtin_shufflevector(b0, b1,
            0, 1, 2, 3, 4, 5, 6, 7, 8, 9, 10, 11, 12, 13, 14, 15);
        acc[t] = __builtin_amdgcn_wmma_f32_16x16x32_bf16(
            /*neg_a=*/false, A, /*neg_b=*/false, Bm,
            /*c_mod=*/(short)0, acc[t], /*reuse_a=*/false, /*reuse_b=*/false);
      }
    }

    // Epilogue for this n: leaky_relu, sigmoid interpolation, product.
#pragma unroll
    for (int t = 0; t < 4; ++t) {
      const float x  = nandw[cIdx[t] * N_VECS + n];
      const float s  = 1.0f / (1.0f + __expf(-x));   // sigmoid
      const float c1 = 2.0f * s - 1.0f;
      const float c0 = 1.0f - s;                     // interp = c0 + c1*cs
#pragma unroll
      for (int r = 0; r < 8; ++r) {
        float cs = acc[t][r];
        cs = cs > 0.0f ? cs : 0.01f * cs;            // leaky_relu
        prod[t][r] *= fmaf(c1, cs, c0);
      }
    }

    // All waves done reading stage n; publish stage n+1.
    __syncthreads();
    if (n < N_VECS - 1) {
#pragma unroll
      for (int i = 0; i < 8; ++i) *(bf16x8*)(wDst + i * 32) = pre[i];
      __syncthreads();
    }
  }

  // C/D layout: VGPR r, lanes 0-15 -> M=r, lanes 16-31 -> M=r+8; N = mLane.
#pragma unroll
  for (int t = 0; t < 4; ++t) {
#pragma unroll
    for (int r = 0; r < 8; ++r) {
      const int ob = (bTile << 4) + (hi << 3) + r;
      out[(size_t)ob * N_CHOICES + cIdx[t]] = prod[t][r];
    }
  }
}

// ---------------------------------------------------------------------------
// Launch
// ---------------------------------------------------------------------------
extern "C" void kernel_launch(void* const* d_in, const int* in_sizes, int n_in,
                              void* d_out, int out_size, void* d_ws, size_t ws_size,
                              hipStream_t stream) {
  const float* query  = (const float*)d_in[0];   // [2048, 2048] f32
  const float* weight = (const float*)d_in[1];   // [2048, 2048] f32
  const float* nandw  = (const float*)d_in[2];   // [2048, 8]    f32
  float* out = (float*)d_out;                    // [2048, 2048] f32

  // Workspace: normalized bf16 copies (8 MB each).
  __bf16* qb = (__bf16*)d_ws;
  __bf16* wb = qb + (size_t)BATCH * ROW_ELEMS;

  // Pass 1: (BATCH + N_CHOICES) * N_VECS = 32768 vectors, one wave32 each.
  {
    const int totalWaves = (BATCH + N_CHOICES) * N_VECS;  // 32768
    const int threads = 256;
    const int blocks = totalWaves * 32 / threads;         // 4096
    nand_norm_bf16<<<blocks, threads, 0, stream>>>(query, weight, qb, wb);
  }

  // Pass 2: (BATCH/128) * (N_CHOICES/64) = 512 workgroups of 8 waves.
  {
    const int blocks = (BATCH / 128) * (N_CHOICES / COLS_PER_WG); // 512
    nand_wmma_kernel<<<blocks, 256, 0, stream>>>(qb, wb, nandw, out);
  }
}
